// MultiHeadAttention_12876311953621
// MI455X (gfx1250) — matrix operand order repaired and checked by reading, not compiled, not run on hardware
//
#include <hip/hip_runtime.h>

// ---------------------------------------------------------------------------
// MultiHeadAttention for MI455X (gfx1250), wave32 + v_wmma_f32_16x16x32_bf16.
// B=2, Lq=Lk=2048, E=512, H=8, dk=64, dv=512, TOP_K=64.
// All GEMM B-operands are pre-packed into WMMA-native bf16 fragment layout so
// the inner loop is pure {b128 loads + cvt_pk_bf16 + v_wmma}.
// ---------------------------------------------------------------------------

#define H_    8
#define DK    64
#define DV    512
#define LQ    2048
#define LK    2048
#define NB    2
#define TOPK  64
#define NEGF  (-3.402823466e38f)

typedef __bf16 bf16_t;
typedef __attribute__((ext_vector_type(16))) __bf16 v16bf;
typedef __attribute__((ext_vector_type(8)))  float  v8f;

__device__ __forceinline__ unsigned fmap_f32(float f) {
  unsigned u = __float_as_uint(f);
  return u ^ ((unsigned)((int)u >> 31) | 0x80000000u);
}

// A fragment (16x32 bf16) from K-contiguous f32 rows, ISA 7.12.2 layout:
// lane<16: M=lane, K={kb..kb+7, 16+kb..16+kb+7} with kb=0 ; lane>=16: kb=8.
__device__ __forceinline__ v16bf a_frag(const float* __restrict__ row, int kb) {
  float4 x0 = *(const float4*)(row + kb);
  float4 x1 = *(const float4*)(row + kb + 4);
  float4 x2 = *(const float4*)(row + kb + 16);
  float4 x3 = *(const float4*)(row + kb + 20);
  v16bf a;
  a[0]=(bf16_t)x0.x; a[1]=(bf16_t)x0.y; a[2]=(bf16_t)x0.z; a[3]=(bf16_t)x0.w;
  a[4]=(bf16_t)x1.x; a[5]=(bf16_t)x1.y; a[6]=(bf16_t)x1.z; a[7]=(bf16_t)x1.w;
  a[8]=(bf16_t)x2.x; a[9]=(bf16_t)x2.y; a[10]=(bf16_t)x2.z; a[11]=(bf16_t)x2.w;
  a[12]=(bf16_t)x3.x; a[13]=(bf16_t)x3.y; a[14]=(bf16_t)x3.z; a[15]=(bf16_t)x3.w;
  return a;
}

// ---------------------------------------------------------------------------
// packB: B(k,n) = base[k*sK + n*sN] (f32, any layout) -> WMMA-native bf16 tiles.
// Fragment (z, nt, ks): lane<16: N=nt*16+lane, K=ks*32+{0..7,16..23};
//                       lane>=16: N=nt*16+lane-16, K=ks*32+{8..15,24..31}.
// Flat thread id == fragment-lane id; output offset = id*16 bf16 (32B/lane).
// ---------------------------------------------------------------------------
__global__ void pack_b_bf16(const float* __restrict__ B, bf16_t* __restrict__ out,
                            int Ks, int Nt, long long sK, long long sN,
                            long long bB, long long bH, long long total) {
  long long id = (long long)blockIdx.x * blockDim.x + threadIdx.x;
  if (id >= total) return;
  int lane = (int)(id & 31);
  long long fid = id >> 5;
  int ks = (int)(fid % Ks); fid /= Ks;
  int nt = (int)(fid % Nt); fid /= Nt;
  int z  = (int)fid;
  const float* base = B + (size_t)(z >> 3) * bB + (size_t)(z & 7) * bH;
  long long n  = nt * 16 + (lane & 15);
  int kb = (lane >> 4) << 3;
  v16bf f;
#pragma unroll
  for (int e = 0; e < 16; ++e)
    f[e] = (bf16_t)base[(long long)(ks * 32 + kb + e + (e >= 8 ? 8 : 0)) * sK + n * sN];
  *(v16bf*)(out + (size_t)id * 16) = f;
}

// ---------------------------------------------------------------------------
// GEMM: C = A(f32,[M,K],ldA, K-contiguous) * Bpk(packed bf16) -> C(f32,ldC)
// z-batched (z = b*8 + h). 8 waves/block; block tile 64(M) x 256(N);
// wave tile 32x64 -> 8 WMMAs per K-step, only b128-class loads in the loop.
// ---------------------------------------------------------------------------
__global__ void wmma_gemm_pk(const float* __restrict__ A, const bf16_t* __restrict__ Bpk,
                             float* __restrict__ C, int M, int N, int K,
                             int ldA, int ldC,
                             long long aB, long long aH, long long cB, long long cH) {
  int z = blockIdx.z;
  A += (size_t)(z >> 3) * aB + (size_t)(z & 7) * aH;
  C += (size_t)(z >> 3) * cB + (size_t)(z & 7) * cH;

  int lane = threadIdx.x & 31;
  int wid  = threadIdx.x >> 5;          // 0..7
  int mw   = wid >> 2, nw = wid & 3;
  int m_base = blockIdx.y * 64 + mw * 32;
  int n_base = blockIdx.x * 256 + nw * 64;
  if (m_base >= M || n_base >= N) return;    // uniform per wave

  int Ks = K >> 5, Nt = N >> 4;
  const bf16_t* bbase =
      Bpk + ((size_t)((size_t)z * Nt + (n_base >> 4)) * Ks) * 512 + lane * 16;
  const float* arow0 = A + (size_t)(m_base + (lane & 15)) * ldA;
  const float* arow1 = arow0 + (size_t)16 * ldA;
  int kb = (lane >> 4) << 3;

  v8f acc[2][4] = {};
  for (int ks = 0; ks < Ks; ++ks) {
    int k0 = ks << 5;
    v16bf a0 = a_frag(arow0 + k0, kb);
    v16bf a1 = a_frag(arow1 + k0, kb);
    const bf16_t* bptr = bbase + (size_t)ks * 512;
#pragma unroll
    for (int j = 0; j < 4; ++j) {
      v16bf b = *(const v16bf*)(bptr + (size_t)j * Ks * 512);
      acc[0][j] = __builtin_amdgcn_wmma_f32_16x16x32_bf16(false, a0, false, b,
                                                          (short)0, acc[0][j], false, false);
      acc[1][j] = __builtin_amdgcn_wmma_f32_16x16x32_bf16(false, a1, false, b,
                                                          (short)0, acc[1][j], false, false);
    }
  }

  int row0  = ((lane >> 4) << 3);
  int col_l = lane & 15;
#pragma unroll
  for (int mi = 0; mi < 2; ++mi)
#pragma unroll
    for (int j = 0; j < 4; ++j) {
      int col = n_base + 16 * j + col_l;
#pragma unroll
      for (int r = 0; r < 8; ++r)
        C[(size_t)(m_base + mi * 16 + row0 + r) * ldC + col] = acc[mi][j][r];
    }
}

// ---------------------------------------------------------------------------
// Attention core: per block (b, h, 16 query rows).
//  1) logits strip [16 x 2048] via WMMA into 128 KB dynamic LDS (scaled+mask)
//  2) per row: 4-pass radix select of the top-64 threshold (skip if qmask<=0.5)
//  3) per row: softmax -> attn_probs f32 in d_out region 2
// ---------------------------------------------------------------------------
__global__ void attn_softmax_kernel(const float*  __restrict__ qp,   // [B,Lq,H,dk] f32
                                    const bf16_t* __restrict__ kpk,  // packed K^T per z
                                    const int*    __restrict__ pmask,// [B,1,Lq,Lk]
                                    const float*  __restrict__ qmask,// [B,Lq]
                                    float* __restrict__ attn) {      // [B,H,Lq,Lk]
  extern __shared__ float slog[];              // [16][LK]
  __shared__ unsigned hist[256];
  __shared__ float    red[256];
  __shared__ unsigned stat[2];

  int q0  = blockIdx.x * 16;
  int h   = blockIdx.y;
  int b   = blockIdx.z;
  int z   = b * H_ + h;
  int tid = threadIdx.x, lane = tid & 31, wid = tid >> 5;
  const float scale = 0.125f;                  // dk^-0.5

  // Q fragments: 16 rows x dk=64 -> two 16x32 bf16 fragments (K-contiguous f32).
  const float* arow = qp + ((size_t)(b * LQ + q0 + (lane & 15)) * H_ + h) * DK;
  int kb = (lane >> 4) << 3;
  v16bf a0 = a_frag(arow, kb);
  v16bf a1 = a_frag(arow + 32, kb);

  // Packed K^T fragments: z batch, Nt = LK/16 tiles, Ks = DK/32 = 2 k-steps.
  const bf16_t* bbase = kpk + ((size_t)z * (LK / 16)) * 2 * 512 + lane * 16;

  for (int t = wid; t < LK / 16; t += 8) {
    const bf16_t* bptr = bbase + (size_t)t * 2 * 512;
    v16bf b0 = *(const v16bf*)(bptr);
    v16bf b1 = *(const v16bf*)(bptr + 512);
    v8f acc = {};
    acc = __builtin_amdgcn_wmma_f32_16x16x32_bf16(false, a0, false, b0, (short)0, acc, false, false);
    acc = __builtin_amdgcn_wmma_f32_16x16x32_bf16(false, a1, false, b1, (short)0, acc, false, false);

    int n    = t * 16 + (lane & 15);
    int row0 = (lane >> 4) << 3;
#pragma unroll
    for (int r = 0; r < 8; ++r) {
      int qrow = q0 + row0 + r;
      int mk = pmask[(size_t)(b * LQ + qrow) * LK + n];
      slog[(row0 + r) * LK + n] = mk ? acc[r] * scale : NEGF;
    }
  }
  __syncthreads();

  for (int r = 0; r < 16; ++r) {
    float* rowp = slog + r * LK;
    float  qm   = qmask[b * LQ + q0 + r];      // uniform across block
    unsigned thresh = 0;

    if (qm > 0.5f) {
      unsigned prefix = 0, pm = 0; int kk = TOPK;
      for (int shift = 24; shift >= 0; shift -= 8) {
        if (tid < 256) hist[tid] = 0;
        __syncthreads();
        for (int i = tid; i < LK; i += blockDim.x) {
          unsigned u = fmap_f32(rowp[i]);
          if ((u & pm) == prefix) atomicAdd(&hist[(u >> shift) & 255], 1u);
        }
        __syncthreads();
        if (tid == 0) {
          int acc2 = 0, bsel = 0;
          for (int bb = 255; bb >= 0; --bb) {
            int c = (int)hist[bb];
            if (acc2 + c >= kk) { bsel = bb; break; }
            acc2 += c;
          }
          stat[0] = prefix | ((unsigned)bsel << shift);
          stat[1] = (unsigned)(kk - acc2);
        }
        __syncthreads();
        prefix = stat[0]; kk = (int)stat[1]; pm |= (255u << shift);
        __syncthreads();
      }
      thresh = prefix;                          // keep iff fmap(v) >= thresh
    }

    float lm = NEGF;
    for (int i = tid; i < LK; i += blockDim.x) {
      float v = rowp[i];
      if (fmap_f32(v) < thresh) v = NEGF;
      rowp[i] = v;
      lm = fmaxf(lm, v);
    }
    red[tid] = lm; __syncthreads();
    for (int s = 128; s > 0; s >>= 1) { if (tid < s) red[tid] = fmaxf(red[tid], red[tid + s]); __syncthreads(); }
    float rmax = red[0]; __syncthreads();

    float ls = 0.f;
    for (int i = tid; i < LK; i += blockDim.x) ls += expf(rowp[i] - rmax);
    red[tid] = ls; __syncthreads();
    for (int s = 128; s > 0; s >>= 1) { if (tid < s) red[tid] += red[tid + s]; __syncthreads(); }
    float inv = 1.0f / red[0]; __syncthreads();

    float* orow = attn + ((size_t)(b * H_ + h) * LQ + q0 + r) * LK;
    for (int i = tid; i < LK; i += blockDim.x) orow[i] = expf(rowp[i] - rmax) * inv;
    __syncthreads();
  }
}

// ---------------------------------------------------------------------------
extern "C" void kernel_launch(void* const* d_in, const int* in_sizes, int n_in,
                              void* d_out, int out_size, void* d_ws, size_t ws_size,
                              hipStream_t stream) {
  const float* q     = (const float*)d_in[0];
  const float* k     = (const float*)d_in[1];
  const float* v     = (const float*)d_in[2];
  const int*   pmask = (const int*)  d_in[3];
  const float* qmask = (const float*)d_in[4];
  const float* Wq    = (const float*)d_in[5];
  const float* Wk    = (const float*)d_in[6];
  const float* Wv    = (const float*)d_in[7];
  const float* Wfc   = (const float*)d_in[8];

  float* out  = (float*)d_out;                           // [B,Lq,DV]
  float* attn = out + (size_t)NB * LQ * DV;              // [B,H,Lq,Lk]

  // Workspace: qp 8MB | kp 8MB | vp 64MB | mixed 64MB | pack region 32MB
  float*  qp    = (float*)d_ws;                          // [B,Lq,H,dk]  = [4096,512]
  float*  kp    = qp + (size_t)(NB * LQ) * (H_ * DK);
  float*  vp    = kp + (size_t)(NB * LK) * (H_ * DK);    // [B,Lk,H,dv]  = [4096,4096]
  float*  mixed = vp + (size_t)(NB * LK) * (H_ * DV);    // [B,Lq,H,dv]  = [4096,4096]
  bf16_t* pkbuf = (bf16_t*)(mixed + (size_t)(NB * LQ) * (H_ * DV));  // reused per GEMM

  dim3 blk(256);

  // ---- Q projection: q[4096,512] @ Wq[512,512] -> qp --------------------
  {
    long long tot = (long long)(512 / 16) * (512 / 32) * 32;          // 16384
    pack_b_bf16<<<dim3((unsigned)(tot / 256)), blk, 0, stream>>>(
        Wq, pkbuf, 512 / 32, 512 / 16, 512, 1, 0, 0, tot);
    wmma_gemm_pk<<<dim3(512 / 256, 4096 / 64, 1), blk, 0, stream>>>(
        q, pkbuf, qp, 4096, 512, 512, 512, 512, 0, 0, 0, 0);
  }
  // ---- K projection ------------------------------------------------------
  {
    long long tot = (long long)(512 / 16) * (512 / 32) * 32;
    pack_b_bf16<<<dim3((unsigned)(tot / 256)), blk, 0, stream>>>(
        Wk, pkbuf, 512 / 32, 512 / 16, 512, 1, 0, 0, tot);
    wmma_gemm_pk<<<dim3(2, 64, 1), blk, 0, stream>>>(
        k, pkbuf, kp, 4096, 512, 512, 512, 512, 0, 0, 0, 0);
  }
  // ---- V projection: v[4096,512] @ Wv[512,4096] -> vp --------------------
  {
    long long tot = (long long)(4096 / 16) * (512 / 32) * 32;         // 131072
    pack_b_bf16<<<dim3((unsigned)(tot / 256)), blk, 0, stream>>>(
        Wv, pkbuf, 512 / 32, 4096 / 16, 4096, 1, 0, 0, tot);
    wmma_gemm_pk<<<dim3(4096 / 256, 64, 1), blk, 0, stream>>>(
        v, pkbuf, vp, 4096, 4096, 512, 512, 4096, 0, 0, 0, 0);
  }
  // ---- Attention: pack K^T per (b,h), then logits+topk+softmax ----------
  {
    // B(k,n) = kp[((b*LK+n)*8+h)*64 + k] : sK=1, sN=512, bB=LK*512, bH=64
    long long tot = (long long)(NB * H_) * (LK / 16) * (DK / 32) * 32;  // 131072
    pack_b_bf16<<<dim3((unsigned)(tot / 256)), blk, 0, stream>>>(
        kp, pkbuf, DK / 32, LK / 16, 1, 512, (long long)LK * 512, 64, tot);
    size_t shmem = (size_t)16 * LK * sizeof(float);                   // 128 KB
    hipFuncSetAttribute((const void*)attn_softmax_kernel,
                        hipFuncAttributeMaxDynamicSharedMemorySize, (int)shmem);
    attn_softmax_kernel<<<dim3(LQ / 16, H_, NB), blk, shmem, stream>>>(
        qp, pkbuf, pmask, qmask, attn);
  }
  // ---- mixed = attn @ v_proj, z = b*8+h ---------------------------------
  {
    // B(k,n) = vp[((b*LK+k)*8+h)*512 + n] : sK=4096, sN=1, bB=LK*4096, bH=512
    long long tot = (long long)(NB * H_) * (512 / 16) * (LK / 32) * 32; // 1048576
    pack_b_bf16<<<dim3((unsigned)(tot / 256)), blk, 0, stream>>>(
        vp, pkbuf, LK / 32, 512 / 16, 4096, 1, (long long)LK * 4096, 512, tot);
    wmma_gemm_pk<<<dim3(512 / 256, 2048 / 64, NB * H_), blk, 0, stream>>>(
        attn, pkbuf, mixed, 2048, 512, 2048, 2048, 4096,
        (long long)H_ * LQ * LK, (long long)LQ * LK,   // A offsets (b,h)
        (long long)LQ * (H_ * DV), (long long)DV);     // C offsets
  }
  // ---- Output FC: mixed[4096,4096] @ Wfc[4096,512] -> d_out -------------
  {
    long long tot = (long long)(512 / 16) * (4096 / 32) * 32;         // 131072
    pack_b_bf16<<<dim3((unsigned)(tot / 256)), blk, 0, stream>>>(
        Wfc, pkbuf, 4096 / 32, 512 / 16, 512, 1, 0, 0, tot);
    wmma_gemm_pk<<<dim3(2, 64, 1), blk, 0, stream>>>(
        mixed, pkbuf, out, 4096, 512, 4096, 4096, 512, 0, 0, 0, 0);
  }
}
